// HGRN2VisionBlock_20048907337744
// MI455X (gfx1250) — matrix operand order repaired and checked by reading, not compiled, not run on hardware
//
#include <hip/hip_runtime.h>
#include <hip/hip_bf16.h>
#include <math.h>

typedef __bf16 bf16_t;
typedef __attribute__((ext_vector_type(16))) __bf16 v16bf;
typedef __attribute__((ext_vector_type(8)))  __bf16 v8bf;
typedef __attribute__((ext_vector_type(8)))  float  v8f;
typedef __attribute__((ext_vector_type(4)))  unsigned int u32x4;
typedef __attribute__((ext_vector_type(8)))  unsigned int u32x8;

#define BDIM   1024
#define SEQ    1024
#define BATCH  16
#define NHEAD  8
#define EDIM   128
#define DVDIM  128
#define MLPDIM 4096
#define TOK    (BATCH*SEQ)   /* 16384 tokens */
#define LN_EPS 1e-6f

// ---------------- device math helpers ----------------
__device__ __forceinline__ float sigmoidf_(float x){ return 1.0f/(1.0f+__expf(-x)); }
__device__ __forceinline__ float logsigmoidf_(float x){
  return fminf(x, 0.0f) - log1pf(__expf(-fabsf(x)));
}
__device__ __forceinline__ float gelu_erf_(float x){
  return 0.5f*x*(1.0f+erff(x*0.70710678118654752f));
}

// ---------------- Tensor Data Mover: 2D tile -> LDS ----------------
// Builds a D# (groups 0/1 per CDNA5 ISA 8.3/8.4) and issues tensor_load_to_lds.
// data_size = 2 bytes (bf16). LDS padding: after pad_interval DWORDs written,
// skip pad_amount DWORDs (codes: interval 3 -> 16 DW = 64B row, amount 3 -> 4 DW = 8 halfs).
__device__ __forceinline__ void tdm_load_tile_bf16(
    unsigned lds_off, const bf16_t* gptr,
    unsigned tile_d0, unsigned tile_d1,
    unsigned tensor_d0, unsigned tensor_d1, unsigned stride0,
    unsigned pad_interval_code, unsigned pad_amount_code)
{
  union { unsigned long long u64; unsigned u32[2]; } ga;
  ga.u64 = (unsigned long long)(uintptr_t)gptr;
  u32x4 g0;
  g0[0] = 1u;                                            // count=1, user D#
  g0[1] = lds_off;                                       // lds_addr
  g0[2] = ga.u32[0];                                     // global_addr[31:0]
  g0[3] = (ga.u32[1] & 0x01FFFFFFu) | (2u << 30);        // global_addr[56:32] | type=2
  u32x8 g1;
  g1[0] = (1u << 16)                                     // data_size = 2 bytes
        | (1u << 20)                                     // pad_enable
        | (pad_interval_code << 22)
        | (pad_amount_code << 25);
  g1[1] = (tensor_d0 & 0xFFFFu) << 16;                   // tensor_dim0[15:0]
  g1[2] = (tensor_d0 >> 16) | ((tensor_d1 & 0xFFFFu) << 16);
  g1[3] = (tensor_d1 >> 16) | (tile_d0 << 16);           // tile_dim0
  g1[4] = tile_d1;                                       // tile_dim1 (tile_dim2=0)
  g1[5] = stride0;                                       // tensor_dim0_stride[31:0]
  g1[6] = 0u;
  g1[7] = 0u;
  u32x4 z = {0u, 0u, 0u, 0u};
  asm volatile("tensor_load_to_lds %0, %1, %2, %3"
               :: "s"(g0), "s"(g1), "s"(z), "s"(z) : "memory");
}

// ---------------- LayerNorm (fp32 in -> bf16 out) ----------------
__global__ __launch_bounds__(256) void ln_kernel(const float* __restrict__ x,
    const float* __restrict__ w, const float* __restrict__ b,
    bf16_t* __restrict__ out, int D)
{
  const int row = blockIdx.x;
  const float* xr = x + (size_t)row * D;
  float s = 0.f, s2 = 0.f;
  for (int i = threadIdx.x; i < D; i += 256) { float v = xr[i]; s += v; s2 += v*v; }
  for (int o = 16; o; o >>= 1) { s += __shfl_down(s, o, 32); s2 += __shfl_down(s2, o, 32); }
  __shared__ float rs[8], rs2[8];
  const int wv = threadIdx.x >> 5;
  if ((threadIdx.x & 31) == 0) { rs[wv] = s; rs2[wv] = s2; }
  __syncthreads();
  s = 0.f; s2 = 0.f;
  for (int i = 0; i < 8; ++i) { s += rs[i]; s2 += rs2[i]; }
  const float mu  = s / (float)D;
  const float var = s2 / (float)D - mu*mu;
  const float inv = rsqrtf(var + LN_EPS);
  bf16_t* orow = out + (size_t)row * D;
  for (int i = threadIdx.x; i < D; i += 256)
    orow[i] = (bf16_t)((xr[i]-mu)*inv*w[i] + b[i]);
}

// -------- weight convert fp32 [K,N] -> bf16 transposed [N,K] --------
__global__ __launch_bounds__(256) void convT_kernel(const float* __restrict__ W,
    bf16_t* __restrict__ WT, int K, int N)
{
  int i = blockIdx.x * 256 + threadIdx.x;
  if (i >= K*N) return;
  int k = i / N, n = i - k*N;
  WT[(size_t)n*K + k] = (bf16_t)W[i];
}

// ---------------- WMMA bf16 GEMM, TDM-staged, fused epilogues ----------------
// C[M,N] = A[M,K] @ B[K,N], B supplied transposed (BT[N,K] row-major).
// Double-buffered LDS tiles filled by the Tensor Data Mover (TENSORcnt),
// issued by wave 0 one iteration ahead so the DMA overlaps the WMMA chain.
// mode 0: out bf16                   (v projection)
// mode 1: silu -> bf16               (q projection)
// mode 2: k=1-sigmoid bf16, g=logsig fp32 (f projection)
// mode 3: +bias, exact GELU -> bf16  (MLP up)
// mode 4: +addsrc -> fp32            (Wo + residual -> h)
// mode 5: +bias +addsrc -> fp32      (MLP down + h -> final out)
#define GBM 128
#define GBN 128
#define GBK 32
#define LPAD 8

__global__ __launch_bounds__(256) void gemm_wmma_kernel(
    const bf16_t* __restrict__ A,
    const bf16_t* __restrict__ BT,
    int M, int N, int K, int mode,
    const float* __restrict__ bias,
    const float* __restrict__ addsrc,
    bf16_t* __restrict__ outB,
    float* __restrict__ outF)
{
  __shared__ __align__(16) bf16_t As[2][GBM][GBK+LPAD];
  __shared__ __align__(16) bf16_t Bs[2][GBN][GBK+LPAD];

  const int tid   = threadIdx.x;
  const int lane  = tid & 31;
  const int wv    = tid >> 5;
  const int mBase = blockIdx.y * GBM;
  const int nBase = blockIdx.x * GBN;

  const bf16_t* aTile = A  + (size_t)mBase * K;
  const bf16_t* bTile = BT + (size_t)nBase * K;

  v8f acc[8];
  {
    v8f z = {};
#pragma unroll
    for (int j = 0; j < 8; ++j) acc[j] = z;
  }

  const int lr  = lane & 15;
  const int khA = (lane & 16) ? 8  : 0;   // 16-bit A 16x32 lane layout
  const int khB = (lane & 16) ? 8 : 0;    // 16-bit B 32x16 lane layout

  const int nSteps = K / GBK;

  // Prologue: DMA tile 0 into buffer 0.
  if (wv == 0) {
    tdm_load_tile_bf16((unsigned)(uintptr_t)&As[0][0][0], aTile,
                       GBK, GBM, (unsigned)K, (unsigned)(M - mBase), (unsigned)K, 3u, 3u);
    tdm_load_tile_bf16((unsigned)(uintptr_t)&Bs[0][0][0], bTile,
                       GBK, GBN, (unsigned)K, (unsigned)(N - nBase), (unsigned)K, 3u, 3u);
  }

  for (int s = 0; s < nSteps; ++s) {
    const int cur = s & 1;
    const int nxt = cur ^ 1;
    if (wv == 0) __builtin_amdgcn_s_wait_tensorcnt((short)0);   // buffer `cur` ready
    __syncthreads();

    // Kick the DMA for the next tile; it runs while we do the WMMA chain below.
    if (wv == 0 && (s + 1) < nSteps) {
      const int kt = (s + 1) * GBK;
      tdm_load_tile_bf16((unsigned)(uintptr_t)&As[nxt][0][0], aTile + kt,
                         GBK, GBM, (unsigned)(K - kt), (unsigned)(M - mBase), (unsigned)K, 3u, 3u);
      tdm_load_tile_bf16((unsigned)(uintptr_t)&Bs[nxt][0][0], bTile + kt,
                         GBK, GBN, (unsigned)(K - kt), (unsigned)(N - nBase), (unsigned)K, 3u, 3u);
    }

    const int arow = wv*16 + lr;
    v8bf alo = *(const v8bf*)&As[cur][arow][khA];
    v8bf ahi = *(const v8bf*)&As[cur][arow][16+khA];
    v16bf af = __builtin_shufflevector(alo, ahi, 0,1,2,3,4,5,6,7,8,9,10,11,12,13,14,15);

    // Load all B fragments first (one dscnt wait), then WMMAs issue back-to-back.
    v16bf bfr[8];
#pragma unroll
    for (int j = 0; j < 8; ++j) {
      const int bcol = j*16 + lr;
      v8bf blo = *(const v8bf*)&Bs[cur][bcol][khB];
      v8bf bhi = *(const v8bf*)&Bs[cur][bcol][16+khB];
      bfr[j] = __builtin_shufflevector(blo, bhi, 0,1,2,3,4,5,6,7,8,9,10,11,12,13,14,15);
    }
#pragma unroll
    for (int j = 0; j < 8; ++j)
      acc[j] = __builtin_amdgcn_wmma_f32_16x16x32_bf16(
          false, af, false, bfr[j], (short)0, acc[j], false, false);

    __syncthreads();   // everyone done with `cur` before it gets overwritten
  }

  // Epilogue: 16x16 f32 C layout -> row = wv*16 + r (+8 for lanes 16..31), col = lane&15
  const int rowOff = (lane & 16) ? 8 : 0;
#pragma unroll
  for (int j = 0; j < 8; ++j) {
    const int col = nBase + j*16 + lr;
#pragma unroll
    for (int r = 0; r < 8; ++r) {
      const int row = mBase + wv*16 + r + rowOff;
      const size_t idx = (size_t)row * N + col;
      const float val = acc[j][r];
      switch (mode) {
        case 0: outB[idx] = (bf16_t)val; break;
        case 1: outB[idx] = (bf16_t)(val * sigmoidf_(val)); break;
        case 2: {
          outB[idx] = (bf16_t)(1.0f - sigmoidf_(val));
          outF[idx] = logsigmoidf_(val);
        } break;
        case 3: outB[idx] = (bf16_t)gelu_erf_(val + bias[col]); break;
        case 4: outF[idx] = val + addsrc[idx]; break;
        case 5: outF[idx] = val + bias[col] + addsrc[idx]; break;
      }
    }
  }
}

// ---------------- HGRN2 gated linear-attention scan ----------------
// One block per (batch, head, direction). State S[E=128, DV=128] fp32 lives in
// registers: thread t owns column d = t&127, e-range [ (t>>7)*64, +64 ).
// Register double-buffering: step l+1's operands are fetched from global while
// the FMA chain for step l runs out of LDS (hides load latency in the scan).
__global__ __launch_bounds__(256) void hgrn2_scan_kernel(
    const bf16_t* __restrict__ q, const bf16_t* __restrict__ k,
    const bf16_t* __restrict__ v, const float* __restrict__ g,
    const float* __restrict__ gnw,
    bf16_t* __restrict__ outFwd, bf16_t* __restrict__ outBwd)
{
  const int dir = blockIdx.x & 1;
  const int bh  = blockIdx.x >> 1;
  const int b   = bh >> 3;     // / NHEAD
  const int h   = bh & 7;
  const int tid = threadIdx.x;
  const int d   = tid & 127;
  const int e0  = (tid >> 7) * 64;

  __shared__ float sq[EDIM], sk[EDIM], seg[EDIM], sv[DVDIM];
  __shared__ float so[256];
  __shared__ float sred[4];

  float S[64];
#pragma unroll
  for (int i = 0; i < 64; ++i) S[i] = 0.f;

  const float scale = 0.08838834764831845f;   // E^-0.5
  bf16_t* outp = dir ? outBwd : outFwd;

  // Preload step 0 into registers.
  float pq = 0.f, pk = 0.f, pg = 0.f, pv = 0.f;
  {
    const int p0 = dir ? (SEQ-1) : 0;
    const size_t off = ((size_t)b*SEQ + p0) * (size_t)BDIM + (size_t)h*128;
    if (tid < 128) { pq = (float)q[off+tid]; pk = (float)k[off+tid]; pg = g[off+tid]; }
    else           { pv = (float)v[off + (tid-128)]; }
  }

  for (int l = 0; l < SEQ; ++l) {
    // Publish staged operands to LDS.
    if (tid < 128) {
      sq[tid]  = pq * scale;
      sk[tid]  = pk;
      seg[tid] = __expf(pg);
    } else {
      sv[tid-128] = pv;
    }
    __syncthreads();

    // Prefetch next step into registers (overlaps the compute below).
    if (l + 1 < SEQ) {
      const int pn = dir ? (SEQ-2-l) : (l+1);
      const size_t off = ((size_t)b*SEQ + pn) * (size_t)BDIM + (size_t)h*128;
      if (tid < 128) { pq = (float)q[off+tid]; pk = (float)k[off+tid]; pg = g[off+tid]; }
      else           { pv = (float)v[off + (tid-128)]; }
    }

    const float vv = sv[d];
    float oacc = 0.f;
#pragma unroll
    for (int i = 0; i < 64; ++i) {
      float Si = S[i]*seg[e0+i] + sk[e0+i]*vv;
      S[i] = Si;
      oacc += sq[e0+i]*Si;
    }
    so[tid] = oacc;
    __syncthreads();

    float o = 0.f;
    if (tid < 128) {
      o = so[tid] + so[tid+128];
      float ss = o*o;
      for (int ofs = 16; ofs; ofs >>= 1) ss += __shfl_down(ss, ofs, 32);
      if ((tid & 31) == 0) sred[tid >> 5] = ss;
    }
    __syncthreads();
    if (tid < 128) {
      const float sumsq = sred[0]+sred[1]+sred[2]+sred[3];
      const float r = rsqrtf(sumsq * (1.0f/128.0f) + LN_EPS);
      outp[((size_t)b*SEQ + l)*BDIM + (size_t)h*128 + tid] = (bf16_t)(o * r * gnw[tid]);
    }
    __syncthreads();
  }
}

// ---------------- merge forward + backward streams ----------------
__global__ __launch_bounds__(256) void merge_kernel(const bf16_t* __restrict__ a,
    const bf16_t* __restrict__ b, bf16_t* __restrict__ m, int n)
{
  int i = blockIdx.x*256 + threadIdx.x;
  if (i < n) m[i] = (bf16_t)((float)a[i] + (float)b[i]);
}

// ---------------- launch ----------------
extern "C" void kernel_launch(void* const* d_in, const int* in_sizes, int n_in,
                              void* d_out, int out_size, void* d_ws, size_t ws_size,
                              hipStream_t stream) {
  (void)in_sizes; (void)n_in; (void)out_size; (void)ws_size;
  const float* hidden = (const float*)d_in[0];
  const float* ln1w   = (const float*)d_in[1];
  const float* ln1b   = (const float*)d_in[2];
  const float* Wq     = (const float*)d_in[3];
  const float* Wf     = (const float*)d_in[4];
  const float* Wi     = (const float*)d_in[5];
  const float* gnw    = (const float*)d_in[6];
  const float* Wo     = (const float*)d_in[7];
  const float* ln2w   = (const float*)d_in[8];
  const float* ln2b   = (const float*)d_in[9];
  const float* W1     = (const float*)d_in[10];
  const float* b1     = (const float*)d_in[11];
  const float* W2     = (const float*)d_in[12];
  const float* b2     = (const float*)d_in[13];
  float* out = (float*)d_out;

  char* ws = (char*)d_ws;
  const size_t MB = 1ull << 20;
  // workspace layout (344 MB total, heavy aliasing of dead buffers)
  bf16_t* xhat = (bf16_t*)(ws + 0);          // 32MB; reused as merged stream m
  bf16_t* qb   = (bf16_t*)(ws + 32*MB);      // 32MB
  bf16_t* kb   = (bf16_t*)(ws + 64*MB);      // 32MB
  bf16_t* vb   = (bf16_t*)(ws + 96*MB);      // 32MB
  float*  gb   = (float*)(ws + 128*MB);      // 64MB fp32 gates
  bf16_t* a1   = (bf16_t*)(ws + 32*MB);      // 128MB alias over q/k/v/g (dead after scan)
  bf16_t* ma   = (bf16_t*)(ws + 192*MB);     // 32MB fwd stream (dead after merge)
  bf16_t* mbuf = (bf16_t*)(ws + 224*MB);     // 32MB bwd stream
  float*  hbuf = (float*)(ws + 256*MB);      // 64MB residual state h
  bf16_t* y2   = (bf16_t*)(ws + 192*MB);     // alias ma: LN2 output
  bf16_t* WqT  = (bf16_t*)(ws + 320*MB);
  bf16_t* WfT  = (bf16_t*)(ws + 322*MB);
  bf16_t* WiT  = (bf16_t*)(ws + 324*MB);
  bf16_t* WoT  = (bf16_t*)(ws + 326*MB);
  bf16_t* W1T  = (bf16_t*)(ws + 328*MB);     // 8MB
  bf16_t* W2T  = (bf16_t*)(ws + 336*MB);     // 8MB
  bf16_t* mm   = xhat;

  // Weights -> bf16, transposed for contiguous K-major tile loads
  int nb = (1024*1024 + 255)/256;
  convT_kernel<<<nb,256,0,stream>>>(Wq, WqT, 1024, 1024);
  convT_kernel<<<nb,256,0,stream>>>(Wf, WfT, 1024, 1024);
  convT_kernel<<<nb,256,0,stream>>>(Wi, WiT, 1024, 1024);
  convT_kernel<<<nb,256,0,stream>>>(Wo, WoT, 1024, 1024);
  nb = (1024*4096 + 255)/256;
  convT_kernel<<<nb,256,0,stream>>>(W1, W1T, 1024, 4096);
  convT_kernel<<<nb,256,0,stream>>>(W2, W2T, 4096, 1024);

  // LN1 -> bf16 activations (shared by both scan directions)
  ln_kernel<<<TOK,256,0,stream>>>(hidden, ln1w, ln1b, xhat, BDIM);

  dim3 g1(1024/GBN, TOK/GBM);
  // q = silu(x@Wq), (k,g) from x@Wf, v = x@Wi
  gemm_wmma_kernel<<<g1,256,0,stream>>>(xhat, WqT, TOK,1024,1024, 1, nullptr, nullptr, qb, nullptr);
  gemm_wmma_kernel<<<g1,256,0,stream>>>(xhat, WfT, TOK,1024,1024, 2, nullptr, nullptr, kb, gb);
  gemm_wmma_kernel<<<g1,256,0,stream>>>(xhat, WiT, TOK,1024,1024, 0, nullptr, nullptr, vb, nullptr);

  // Bidirectional gated linear-attention scan (fused per-head RMSNorm)
  hgrn2_scan_kernel<<<BATCH*NHEAD*2,256,0,stream>>>(qb, kb, vb, gb, gnw, ma, mbuf);

  // cross-merge (sum of normalized streams) -> single Wo GEMM over B*L tokens
  merge_kernel<<<(TOK*1024)/256,256,0,stream>>>(ma, mbuf, mm, TOK*1024);

  // h = residual + m@Wo
  gemm_wmma_kernel<<<g1,256,0,stream>>>(mm, WoT, TOK,1024,1024, 4, nullptr, hidden, nullptr, hbuf);

  // LN2 -> bf16
  ln_kernel<<<TOK,256,0,stream>>>(hbuf, ln2w, ln2b, y2, BDIM);

  // MLP up: a1 = gelu(y@W1 + b1)
  dim3 g2(4096/GBN, TOK/GBM);
  gemm_wmma_kernel<<<g2,256,0,stream>>>(y2, W1T, TOK,4096,1024, 3, b1, nullptr, a1, nullptr);

  // MLP down + final residual: out = h + a1@W2 + b2
  gemm_wmma_kernel<<<g1,256,0,stream>>>(a1, W2T, TOK,1024,4096, 5, b2, hbuf, nullptr, out);
}
